// SelfAttention_89627377533629
// MI455X (gfx1250) — matrix operand order repaired and checked by reading, not compiled, not run on hardware
//
#include <hip/hip_runtime.h>
#include <hip/hip_bf16.h>

typedef __bf16 bf16_t;
typedef __attribute__((ext_vector_type(16))) __bf16 bf16x16;
typedef __attribute__((ext_vector_type(8)))  float  f32x8;
typedef __attribute__((ext_vector_type(4)))  unsigned u32x4;
typedef __attribute__((ext_vector_type(8)))  int      i32x8;
typedef __attribute__((ext_vector_type(4)))  int      i32x4;

static __device__ __forceinline__ bf16_t f2bf(float f) {
  __hip_bfloat16 h = __float2bfloat16(f);
  return __builtin_bit_cast(bf16_t, h);
}
static __device__ __forceinline__ float bf2f(bf16_t b) {
  return __bfloat162float(__builtin_bit_cast(__hip_bfloat16, b));
}

// ---------------------------------------------------------------- fp32 -> bf16
__global__ void cvt_f32_bf16(const float* __restrict__ src,
                             bf16_t* __restrict__ dst, int n4) {
  int i = blockIdx.x * blockDim.x + threadIdx.x;
  if (i >= n4) return;
  float4 v = ((const float4*)src)[i];
  union { ushort4 u; bf16_t h[4]; } o;
  o.h[0] = f2bf(v.x); o.h[1] = f2bf(v.y); o.h[2] = f2bf(v.z); o.h[3] = f2bf(v.w);
  ((ushort4*)dst)[i] = o.u;
}

// ---------------------------------------------------------------- TDM helper
// 2D tensor tile -> LDS via Tensor Data Mover (CDNA5 ISA/8).
// D# group0: count=1, lds_addr, global_addr(57b), type=2.
// D# group1: data_size=2B, optional LDS padding, tensor dims/strides, tile dims.
// This toolchain's builtin takes 6 args (g0, g1, g2, g3, extra, cpol).
static __device__ __forceinline__ void tdm_load_tile_bf16(
    const bf16_t* gptr, unsigned lds_off,
    int tileD0, int tileD1,              // tile: rowElems x rows
    unsigned tensorD0, unsigned tensorD1,// tensor extents (elements)
    unsigned strideD0,                   // row stride (elements)
    int padEnable, int padIntervalCode, int padAmountCode) {
  unsigned long long ga = (unsigned long long)gptr;
  u32x4 g0;
  g0[0] = 1u;                                        // count=1, no gather
  g0[1] = lds_off;                                   // LDS byte address
  g0[2] = (unsigned)(ga & 0xFFFFFFFFull);            // global_addr[31:0]
  g0[3] = (unsigned)((ga >> 32) & 0x01FFFFFFull)     // global_addr[56:32]
        | (2u << 30);                                // type=2 ("image")
  i32x8 g1;
  g1[0] = (1 << 16)                                  // data_size = 2 bytes
        | (padEnable << 20)
        | (padIntervalCode << 22)
        | (padAmountCode << 25);
  g1[1] = (int)((tensorD0 & 0xFFFFu) << 16);         // tensor_dim0[15:0]
  g1[2] = (int)(((tensorD0 >> 16) & 0xFFFFu)
        | ((tensorD1 & 0xFFFFu) << 16));             // dim0 hi | dim1 lo
  g1[3] = (int)(((tensorD1 >> 16) & 0xFFFFu)
        | (((unsigned)tileD0 & 0xFFFFu) << 16));     // dim1 hi | tile_dim0
  g1[4] = (int)((unsigned)tileD1 & 0xFFFFu);         // tile_dim1 (tile_dim2=0)
  g1[5] = (int)strideD0;                             // tensor_dim0_stride lo
  g1[6] = 0;                                         // stride hi | dim1_stride lo
  g1[7] = 0;
  i32x4 gz4 = {0, 0, 0, 0};                          // 2D: groups 2/3 unused
  i32x8 gz8 = {0, 0, 0, 0, 0, 0, 0, 0};
  __builtin_amdgcn_tensor_load_to_lds(g0, g1, gz4, gz4, gz8, 0);
}

// ---------------------------------------------------------------- WMMA GEMM
// C[M,N] = A[M,K] (bf16, row-major) * B[K,N] (bf16, row-major) + bias
// mode 0: store bf16 row-major; mode 1: store f32 row-major;
// mode 2: store bf16 transposed-per-batch as C[b][n][s] (b = m>>11, s = m&2047)
#define BM 128
#define BN 64
#define BK 32
#define LDA_S 34   // BK + 2 pad (bf16) == TDM pad_interval 16 dwords + 1 dword
#define LDB_S 34

__global__ __launch_bounds__(256, 2)
void gemm_wmma_bf16(const bf16_t* __restrict__ A, const bf16_t* __restrict__ B,
                    const float* __restrict__ bias, void* __restrict__ C,
                    int M, int N, int K, int mode) {
  __shared__ bf16_t As[2][BM * LDA_S];  // [buf][row][k], filled by TDM
  __shared__ bf16_t Bs[BN * LDB_S];     // [col][k] (transposed on stage-in)

  const int t     = threadIdx.x;
  const int lane  = t & 31;
  const int wave  = t >> 5;
  const int hl    = lane >> 4;          // lane half (0/1)
  const int ln16  = lane & 15;
  const int wm    = wave & 3;           // 4 waves along M
  const int wn    = wave >> 2;          // 2 waves along N
  const int m0    = blockIdx.y * BM;
  const int n0    = blockIdx.x * BN;

  const int bkk   = t >> 3;             // 0..31 (B staging row)
  const int bnn   = (t & 7) * 8;        // B staging col group

  f32x8 acc[2][2] = {};

  // prologue: async-DMA first A tile into buffer 0 (wave 0 only; EXEC ignored)
  if (wave == 0) {
    tdm_load_tile_bf16(A + (size_t)m0 * K,
                       (unsigned)(unsigned long long)&As[0][0],
                       BK, BM, (unsigned)K, (unsigned)M, (unsigned)K,
                       1, 3 /*16 dwords*/, 0 /*+1 dword*/);
  }

  for (int kb = 0; kb < K; kb += BK) {
    const int buf = (kb >> 5) & 1;
    // issue next A tile DMA into the other buffer (overlaps with compute)
    if (wave == 0 && kb + BK < K) {
      tdm_load_tile_bf16(A + (size_t)m0 * K + kb + BK,
                         (unsigned)(unsigned long long)&As[buf ^ 1][0],
                         BK, BM, (unsigned)K, (unsigned)M, (unsigned)K,
                         1, 3, 0);
    }
    if (kb + BK < K)  // next-tile B prefetch -> global_prefetch_b8
      __builtin_prefetch(B + (size_t)(kb + BK + bkk) * N + n0 + bnn, 0, 1);

    // stage B: 32x64 tile, transposed into Bs[col][k]
    {
      union { uint4 u4; bf16_t h[8]; } v;
      v.u4 = *(const uint4*)(B + (size_t)(kb + bkk) * N + n0 + bnn);
      #pragma unroll
      for (int j = 0; j < 8; ++j) Bs[(bnn + j) * LDB_S + bkk] = v.h[j];
    }
    // current A tile must be resident; next one may stay in flight
    if (wave == 0) {
      if (kb + BK < K) __builtin_amdgcn_s_wait_tensorcnt((short)1);
      else             __builtin_amdgcn_s_wait_tensorcnt((short)0);
    }
    __syncthreads();

    // fragment gathers (ISA 7.12.2 16-bit layouts, wave32)
    union FR { bf16x16 v; unsigned u[8]; } afr[2], bfr[2];
    #pragma unroll
    for (int ti = 0; ti < 2; ++ti) {
      int row = wm * 32 + ti * 16 + ln16;
      const unsigned* base = (const unsigned*)&As[buf][row * LDA_S];
      #pragma unroll
      for (int p = 0; p < 8; ++p) {
        int ks = ((p < 4) ? (2 * p) : (2 * p + 8)) + 8 * hl; // A K-pattern
        afr[ti].u[p] = base[ks >> 1];
      }
    }
    #pragma unroll
    for (int tj = 0; tj < 2; ++tj) {
      int col = wn * 32 + tj * 16 + ln16;
      const unsigned* base = (const unsigned*)&Bs[col * LDB_S];
      #pragma unroll
      for (int p = 0; p < 8; ++p) {
        int ks = ((p < 4) ? (2 * p) : (2 * p + 8)) + 8 * hl; // B K-pattern
        bfr[tj].u[p] = base[ks >> 1];
      }
    }
    __syncthreads();

    #pragma unroll
    for (int ti = 0; ti < 2; ++ti)
      #pragma unroll
      for (int tj = 0; tj < 2; ++tj)
        acc[ti][tj] = __builtin_amdgcn_wmma_f32_16x16x32_bf16(
            false, afr[ti].v, false, bfr[tj].v, (short)0, acc[ti][tj],
            false, false);
  }

  // store: D layout = row (vgpr + 8*half), col (lane%16)
  #pragma unroll
  for (int ti = 0; ti < 2; ++ti) {
    #pragma unroll
    for (int tj = 0; tj < 2; ++tj) {
      int gn = n0 + wn * 32 + tj * 16 + ln16;
      float bb = bias ? bias[gn] : 0.0f;
      #pragma unroll
      for (int r = 0; r < 8; ++r) {
        int gm = m0 + wm * 32 + ti * 16 + hl * 8 + r;
        float val = acc[ti][tj][r] + bb;
        if (mode == 1) {
          ((float*)C)[(size_t)gm * N + gn] = val;
        } else if (mode == 0) {
          ((bf16_t*)C)[(size_t)gm * N + gn] = f2bf(val);
        } else { // mode 2: K^T store, C[b][n][s], seq=2048
          int bi = gm >> 11, sp = gm & 2047;
          ((bf16_t*)C)[((size_t)bi * N + gn) * 2048 + sp] = f2bf(val);
        }
      }
    }
  }
}

// ---------------------------------------------------------------- row softmax
// in-place over bf16 rows of length 2048 (8 elems/thread, 256 threads)
__global__ __launch_bounds__(256)
void softmax_rows_bf16(bf16_t* __restrict__ S, float scale) {
  __shared__ float red[256];
  const int t = threadIdx.x;
  bf16_t* rp = S + (size_t)blockIdx.x * 2048;
  float v[8];
  float lmax = -3.4e38f;
  #pragma unroll
  for (int j = 0; j < 8; ++j) {
    v[j] = bf2f(rp[t + j * 256]) * scale;
    lmax = fmaxf(lmax, v[j]);
  }
  red[t] = lmax; __syncthreads();
  for (int s = 128; s > 0; s >>= 1) {
    if (t < s) red[t] = fmaxf(red[t], red[t + s]);
    __syncthreads();
  }
  float rmax = red[0]; __syncthreads();
  float lsum = 0.f;
  #pragma unroll
  for (int j = 0; j < 8; ++j) { v[j] = __expf(v[j] - rmax); lsum += v[j]; }
  red[t] = lsum; __syncthreads();
  for (int s = 128; s > 0; s >>= 1) {
    if (t < s) red[t] += red[t + s];
    __syncthreads();
  }
  float inv = 1.0f / red[0];
  #pragma unroll
  for (int j = 0; j < 8; ++j) rp[t + j * 256] = f2bf(v[j] * inv);
}

// ---------------------------------------------------------------- launch
extern "C" void kernel_launch(void* const* d_in, const int* in_sizes, int n_in,
                              void* d_out, int out_size, void* d_ws, size_t ws_size,
                              hipStream_t stream) {
  const float* x  = (const float*)d_in[0];
  const float* wq = (const float*)d_in[1];
  const float* bq = (const float*)d_in[2];
  const float* wk = (const float*)d_in[3];
  const float* bk = (const float*)d_in[4];
  const float* wv = (const float*)d_in[5];
  const float* bv = (const float*)d_in[6];
  const float* wo = (const float*)d_in[7];
  const float* bo = (const float*)d_in[8];
  (void)in_sizes; (void)n_in; (void)out_size; (void)ws_size;

  const int Bt = 4, S = 2048, D = 1024;
  const int Mall = Bt * S; // 8192

  char* ws = (char*)d_ws;
  size_t off = 0;
  auto take = [&](size_t bytes) {
    size_t o = off; off += (bytes + 255) & ~(size_t)255; return o;
  };
  bf16_t* xb  = (bf16_t*)(ws + take((size_t)Mall * D * 2));   // 16 MB
  bf16_t* wqb = (bf16_t*)(ws + take((size_t)D * D * 2));      //  2 MB
  bf16_t* wkb = (bf16_t*)(ws + take((size_t)D * D * 2));
  bf16_t* wvb = (bf16_t*)(ws + take((size_t)D * D * 2));
  bf16_t* wob = (bf16_t*)(ws + take((size_t)D * D * 2));
  bf16_t* Qb  = (bf16_t*)(ws + take((size_t)Mall * D * 2));   // 16 MB
  bf16_t* Ktb = (bf16_t*)(ws + take((size_t)Bt * D * S * 2)); // 16 MB [b][d][s]
  bf16_t* Vb  = (bf16_t*)(ws + take((size_t)Mall * D * 2));   // 16 MB
  bf16_t* Sb  = (bf16_t*)(ws + take((size_t)S * S * 2));      //  8 MB, per-batch reuse
  bf16_t* Ob  = (bf16_t*)(ws + take((size_t)Mall * D * 2));   // 16 MB  (~96 MB total)

  // fp32 -> bf16
  cvt_f32_bf16<<<(Mall * D / 4 + 255) / 256, 256, 0, stream>>>(x,  xb,  Mall * D / 4);
  cvt_f32_bf16<<<(D * D / 4 + 255) / 256, 256, 0, stream>>>(wq, wqb, D * D / 4);
  cvt_f32_bf16<<<(D * D / 4 + 255) / 256, 256, 0, stream>>>(wk, wkb, D * D / 4);
  cvt_f32_bf16<<<(D * D / 4 + 255) / 256, 256, 0, stream>>>(wv, wvb, D * D / 4);
  cvt_f32_bf16<<<(D * D / 4 + 255) / 256, 256, 0, stream>>>(wo, wob, D * D / 4);

  dim3 blk(256);
  // Q = x@wq + bq ; K^T = (x@wk + bk)^T per batch ; V = x@wv + bv
  gemm_wmma_bf16<<<dim3(D / BN, Mall / BM), blk, 0, stream>>>(xb, wqb, bq, Qb,  Mall, D, D, 0);
  gemm_wmma_bf16<<<dim3(D / BN, Mall / BM), blk, 0, stream>>>(xb, wkb, bk, Ktb, Mall, D, D, 2);
  gemm_wmma_bf16<<<dim3(D / BN, Mall / BM), blk, 0, stream>>>(xb, wvb, bv, Vb,  Mall, D, D, 0);

  const float scale = 1.0f / 32.0f; // 1/sqrt(1024)
  for (int b = 0; b < Bt; ++b) {
    const bf16_t* Q_b  = Qb  + (size_t)b * S * D;
    const bf16_t* Kt_b = Ktb + (size_t)b * D * S;
    const bf16_t* V_b  = Vb  + (size_t)b * S * D;
    bf16_t*       O_b  = Ob  + (size_t)b * S * D;
    // S = Q @ K^T   [2048 x 2048], K-reduction over D
    gemm_wmma_bf16<<<dim3(S / BN, S / BM), blk, 0, stream>>>(Q_b, Kt_b, nullptr, Sb, S, S, D, 0);
    // P = softmax(S * scale) in place
    softmax_rows_bf16<<<S, 256, 0, stream>>>(Sb, scale);
    // O = P @ V     [2048 x 1024], K-reduction over S
    gemm_wmma_bf16<<<dim3(D / BN, S / BM), blk, 0, stream>>>(Sb, V_b, nullptr, O_b, S, D, S, 0);
  }
  // out = O @ wo + bo (fp32 store)
  gemm_wmma_bf16<<<dim3(D / BN, Mall / BM), blk, 0, stream>>>(Ob, wob, bo, d_out, Mall, D, D, 1);
}
